// GMKEALayer_48799418417675
// MI455X (gfx1250) — matrix operand order repaired and checked by reading, not compiled, not run on hardware
//
#include <hip/hip_runtime.h>
#include <hip/hip_bf16.h>

typedef __attribute__((ext_vector_type(16))) _Float16 v16h;
typedef __attribute__((ext_vector_type(8)))  _Float16 v8h;
typedef __attribute__((ext_vector_type(8)))  float    v8f;

#define N_NODES 8192
#define F_IN    512
#define F_OUT   256
#define NEGINF  -9e15f
#define NCHUNK  4
#define QT      (N_NODES/16)   /* 512 query tiles */

union V16U { v16h v; v8h h[2]; };

static __device__ __forceinline__ v8f wmma_f16(v16h a, v16h b, v8f c) {
  // v_wmma_f32_16x16x32_f16: D = A(16x32 f16) * B(32x16 f16) + C(16x16 f32)
  return __builtin_amdgcn_wmma_f32_16x16x32_f16(false, a, false, b, (short)0, c,
                                                false, false);
}

// ---------------------------------------------------------------------------
// Kernel 1: Wt16[n][k] = (f16) W[k][n]   (256 x 512, W transposed so GEMM1
// B-fragments are contiguous per lane)
// ---------------------------------------------------------------------------
__global__ void k_convW(const float* __restrict__ W, _Float16* __restrict__ Wt16) {
  int idx = blockIdx.x * blockDim.x + threadIdx.x;   // 0 .. 131071
  int n = idx >> 9;        // /512
  int k = idx & 511;
  Wt16[idx] = (_Float16)W[k * F_OUT + n];
}

// ---------------------------------------------------------------------------
// Kernel 2: Wh = h @ W via WMMA.  One wave per 16x16 output tile, K=512.
// Writes Wh16 (row major f16), WhT16 (col major f16) and t1 = Wh[:,1]^2.
// ---------------------------------------------------------------------------
__global__ void k_gemm1(const float* __restrict__ h, const _Float16* __restrict__ Wt16,
                        _Float16* __restrict__ Wh16, _Float16* __restrict__ WhT16,
                        float* __restrict__ t1) {
  const int lane = threadIdx.x;
  const int l15 = lane & 15;
  const int hi  = lane >> 4;
  const int qt = blockIdx.x, nt = blockIdx.y;

  v8f acc = {};
  const float*    hrow = h    + (size_t)(qt * 16 + l15) * F_IN;
  const _Float16* brow = Wt16 + (size_t)(nt * 16 + l15) * F_IN;
#pragma unroll
  for (int kk = 0; kk < F_IN / 32; ++kk) {
    V16U a;
    const float* p0 = hrow + kk * 32 + hi * 16;       // A lane layout: K runs
    const float* p1 = hrow + kk * 32 + hi * 16 + 8;   // [hi*16..+7], [hi*16+8..+7]
#pragma unroll
    for (int i = 0; i < 8; ++i) {
      a.h[0][i] = (_Float16)p0[i];
      a.h[1][i] = (_Float16)p1[i];
    }
    v16h b = *(const v16h*)(brow + kk * 32 + hi * 16);  // K contiguous per lane
    acc = wmma_f16(a.v, b, acc);
  }
  const int col = nt * 16 + l15;
#pragma unroll
  for (int v = 0; v < 8; ++v) {
    int row = qt * 16 + v + 8 * hi;                   // C layout: M = v + 8*hi
    float c = acc[v];
    _Float16 ch = (_Float16)c;
    Wh16[(size_t)row * F_OUT + col]   = ch;
    WhT16[(size_t)col * N_NODES + row] = ch;
    if (col == 1) t1[row] = c * c;                    // reference's column-1 bug
  }
}

// ---------------------------------------------------------------------------
// Kernel 3: fused masked-attention chunk (flash-attention style).
// One wave (32 thr) per (query-tile, key-chunk).  Streams adj once from HBM;
// Wh16/WhT16/t1 are L2-resident.  Produces partial acc[16x256], m[16], l[16].
// ---------------------------------------------------------------------------
__global__ void k_attn(const _Float16* __restrict__ Wh16,
                       const _Float16* __restrict__ WhT16,
                       const float* __restrict__ t1, const int* __restrict__ adj,
                       const float* __restrict__ Mui,
                       float* __restrict__ pacc, float* __restrict__ pm,
                       float* __restrict__ pl) {
  __shared__ __align__(16) _Float16 pTile[16][32];   // P transpose staging (1 KB)

  const int lane = threadIdx.x;
  const int l15 = lane & 15;
  const int hi  = lane >> 4;
  const int qt = blockIdx.x;
  const int ch = blockIdx.y;
  const int qbase = qt * 16;

  // softmax(Mui) -> collapsed kernel coefficients (redundant per thread, cheap)
  float mu[6];
  float mx = Mui[0];
#pragma unroll
  for (int i = 1; i < 6; ++i) mx = fmaxf(mx, Mui[i]);
  float ssum = 0.f;
#pragma unroll
  for (int i = 0; i < 6; ++i) { mu[i] = __expf(Mui[i] - mx); ssum += mu[i]; }
#pragma unroll
  for (int i = 0; i < 6; ++i) mu[i] /= ssum;
  const float aC = mu[0] + mu[1] + mu[2] + mu[3];
  const float bC = 0.5f * (mu[0] + mu[1]) + 1.0f * (mu[2] + mu[3]);
  const float c4 = (float)F_OUT * mu[4];
  const float c5 = (float)F_OUT * mu[5];

  float t1q[8], m[8], l[8];
  v8f acc[16];
  v8f zero = {};
#pragma unroll
  for (int v = 0; v < 8; ++v) {
    t1q[v] = t1[qbase + v + 8 * hi];
    m[v] = NEGINF;
    l[v] = 0.f;
  }
#pragma unroll
  for (int n = 0; n < 16; ++n) acc[n] = zero;

  const int jBeg = ch * (N_NODES / NCHUNK);
  const int jEnd = jBeg + (N_NODES / NCHUNK);
  const _Float16* qrow = Wh16 + (size_t)(qbase + l15) * F_OUT;

  for (int jj = jBeg; jj < jEnd; jj += 32) {
    // ---- S = Q * K^T for two 16-wide key sub-tiles (K = F_OUT = 256) ----
    v8f c0 = zero, c1 = zero;
    const _Float16* b0row = Wh16 + (size_t)(jj + l15) * F_OUT;
    const _Float16* b1row = Wh16 + (size_t)(jj + 16 + l15) * F_OUT;
#pragma unroll
    for (int kk = 0; kk < F_OUT / 32; ++kk) {
      V16U a;
      a.h[0] = *(const v8h*)(qrow + kk * 32 + hi * 16);
      a.h[1] = *(const v8h*)(qrow + kk * 32 + hi * 16 + 8);
      v16h b0 = *(const v16h*)(b0row + kk * 32 + hi * 16);
      v16h b1 = *(const v16h*)(b1row + kk * 32 + hi * 16);
      c0 = wmma_f16(a.v, b0, c0);
      c1 = wmma_f16(a.v, b1, c1);
    }

    // ---- e = a*G + b + c4*exp(-.5*dt) + c5*exp(-dt), mask, online softmax ----
    float t1c0 = t1[jj + l15];
    float t1c1 = t1[jj + 16 + l15];
    float scl[8];
#pragma unroll
    for (int v = 0; v < 8; ++v) {
      int row = qbase + v + 8 * hi;
      const int* arow = adj + (size_t)row * N_NODES + jj;
      float d0 = t1q[v] - t1c0, d1 = t1q[v] - t1c1;
      float e0 = aC * c0[v] + bC + c4 * __expf(-0.5f * d0) + c5 * __expf(-d0);
      float e1 = aC * c1[v] + bC + c4 * __expf(-0.5f * d1) + c5 * __expf(-d1);
      e0 = (arow[l15]      > 0) ? e0 : NEGINF;
      e1 = (arow[16 + l15] > 0) ? e1 : NEGINF;

      float rm = fmaxf(e0, e1);                       // row-max over 16 lanes
#pragma unroll
      for (int off = 1; off < 16; off <<= 1) rm = fmaxf(rm, __shfl_xor(rm, off, 32));
      float mn = fmaxf(m[v], rm);
      float p0 = __expf(e0 - mn), p1 = __expf(e1 - mn);
      float rs = p0 + p1;
#pragma unroll
      for (int off = 1; off < 16; off <<= 1) rs += __shfl_xor(rs, off, 32);
      scl[v] = __expf(m[v] - mn);
      l[v] = l[v] * scl[v] + rs;
      m[v] = mn;
      pTile[v + 8 * hi][l15]      = (_Float16)p0;     // stage P for transpose
      pTile[v + 8 * hi][16 + l15] = (_Float16)p1;
    }
#pragma unroll
    for (int n = 0; n < 16; ++n)
#pragma unroll
      for (int v = 0; v < 8; ++v) acc[n][v] = acc[n][v] * scl[v];

    __syncthreads();                                  // 1-wave WG: ~S_NOP + dscnt wait
    V16U pA;                                          // P as A-fragment (16x32)
    pA.h[0] = *(const v8h*)&pTile[l15][hi * 16];
    pA.h[1] = *(const v8h*)&pTile[l15][hi * 16 + 8];
    __syncthreads();

    // ---- acc += P(16x32) @ V(32x256); B-frags contiguous thanks to WhT16 ----
#pragma unroll
    for (int n = 0; n < 16; ++n) {
      v16h vf = *(const v16h*)(WhT16 + (size_t)(n * 16 + l15) * N_NODES + jj + hi * 16);
      acc[n] = wmma_f16(pA.v, vf, acc[n]);
    }
  }

  // ---- store partials (un-normalized) ----
  float* pa = pacc + (size_t)((ch * QT + qt) * 16) * F_OUT;
#pragma unroll
  for (int n = 0; n < 16; ++n)
#pragma unroll
    for (int v = 0; v < 8; ++v)
      pa[(v + 8 * hi) * F_OUT + n * 16 + l15] = acc[n][v];
  if (l15 == 0) {
    int base = (ch * QT + qt) * 16;
#pragma unroll
    for (int v = 0; v < 8; ++v) {
      pm[base + v + 8 * hi] = m[v];
      pl[base + v + 8 * hi] = l[v];
    }
  }
}

// ---------------------------------------------------------------------------
// Kernel 4: merge NCHUNK partials (log-sum-exp), normalize, ELU, write output.
// One block per row, one thread per output column.
// ---------------------------------------------------------------------------
__global__ void k_merge(const float* __restrict__ pacc, const float* __restrict__ pm,
                        const float* __restrict__ pl, float* __restrict__ out) {
  int row = blockIdx.x;     // 0..8191
  int col = threadIdx.x;    // 0..255
  int qt = row >> 4, rl = row & 15;
  float mv[NCHUNK], lv[NCHUNK];
  float mM = NEGINF;
#pragma unroll
  for (int c = 0; c < NCHUNK; ++c) {
    mv[c] = pm[(c * QT + qt) * 16 + rl];
    lv[c] = pl[(c * QT + qt) * 16 + rl];
    mM = fmaxf(mM, mv[c]);
  }
  float L = 0.f, val = 0.f;
#pragma unroll
  for (int c = 0; c < NCHUNK; ++c) {
    float w = __expf(mv[c] - mM);
    L += lv[c] * w;
    val += w * pacc[(size_t)((c * QT + qt) * 16 + rl) * F_OUT + col];
  }
  val /= L;
  out[(size_t)row * F_OUT + col] = (val > 0.f) ? val : (__expf(val) - 1.f);
}

// ---------------------------------------------------------------------------
extern "C" void kernel_launch(void* const* d_in, const int* in_sizes, int n_in,
                              void* d_out, int out_size, void* d_ws, size_t ws_size,
                              hipStream_t stream) {
  const float* h   = (const float*)d_in[0];   // [8192,512]
  const float* W   = (const float*)d_in[1];   // [512,256]
  const float* Mui = (const float*)d_in[2];   // [6,1]
  const int*   adj = (const int*)d_in[3];     // [8192,8192]
  float* out = (float*)d_out;                 // [8192,256]

  char* ws = (char*)d_ws;
  size_t off = 0;
  _Float16* Wt16  = (_Float16*)(ws + off); off += (size_t)F_OUT * F_IN * 2;      // 256 KB
  _Float16* Wh16  = (_Float16*)(ws + off); off += (size_t)N_NODES * F_OUT * 2;   // 4 MB
  _Float16* WhT16 = (_Float16*)(ws + off); off += (size_t)F_OUT * N_NODES * 2;   // 4 MB
  float* t1   = (float*)(ws + off); off += (size_t)N_NODES * 4;                  // 32 KB
  float* pacc = (float*)(ws + off); off += (size_t)NCHUNK * N_NODES * F_OUT * 4; // 33.5 MB
  float* pm   = (float*)(ws + off); off += (size_t)NCHUNK * N_NODES * 4;
  float* pl   = (float*)(ws + off); off += (size_t)NCHUNK * N_NODES * 4;

  k_convW<<<(F_OUT * F_IN) / 256, 256, 0, stream>>>(W, Wt16);
  k_gemm1<<<dim3(QT, F_OUT / 16), 32, 0, stream>>>(h, Wt16, Wh16, WhT16, t1);
  k_attn<<<dim3(QT, NCHUNK), 32, 0, stream>>>(Wh16, WhT16, t1, adj, Mui, pacc, pm, pl);
  k_merge<<<N_NODES, F_OUT, 0, stream>>>(pacc, pm, pl, out);
}
